// FlashbackAttention_85246510891494
// MI455X (gfx1250) — matrix operand order repaired and checked by reading, not compiled, not run on hardware
//
#include <hip/hip_runtime.h>
#include <hip/hip_bf16.h>

#define HIDDEN 1024
#define SEQ    2048
#define BATCH  32
#define WSTRIDE 2048           // W_proj row stride (H x 2H)

typedef __attribute__((ext_vector_type(16))) __bf16 v16bf;
typedef __attribute__((ext_vector_type(8)))  float  v8f;

__device__ __forceinline__ unsigned short f2bf(float f) {
    __bf16 h = (__bf16)f;
    return __builtin_bit_cast(unsigned short, h);
}

// CDNA5 has a hardware V_TANH_F32 transcendental; use it if the toolchain
// exposes a builtin, else fall back to libm.
__device__ __forceinline__ float fast_tanhf(float x) {
#if __has_builtin(__builtin_amdgcn_tanhf)
    return __builtin_amdgcn_tanhf(x);
#elif __has_builtin(__builtin_amdgcn_tanh_f32)
    return __builtin_amdgcn_tanh_f32(x);
#else
    return tanhf(x);
#endif
}

__device__ __forceinline__ void wait_asynccnt0() {
#if __has_builtin(__builtin_amdgcn_s_wait_asynccnt)
    __builtin_amdgcn_s_wait_asynccnt(0);
#else
    asm volatile("s_wait_asynccnt 0x0" ::: "memory");
#endif
}

// ---------------------------------------------------------------------------
// Kernel 1: pack Wp (= W_proj[:, :H]) as bf16 in WMMA B-fragment order.
// Fragment f = nt*32 + kb covers B[k = kb*32 .. +31][n = nt*16 .. +15],
// B[k][n] = Wp[n][k]. Within a fragment, lane's 16 elements are contiguous:
//   element (lane, 2j+p) = B[K = (lane>=16?16:0) + 2j + p][N = lane&15]
// contiguous K per lane half (same order as the A reads), so the consumer does 2x b128/lane.
// ---------------------------------------------------------------------------
__global__ void pack_wp_kernel(const float* __restrict__ W, unsigned short* __restrict__ wp) {
    int gid  = blockIdx.x * 256 + threadIdx.x;          // 0 .. 1M-1
    int frag = gid >> 9;
    int e    = gid & 511;
    int lane = e >> 4;
    int rem  = e & 15;
    int j    = rem >> 1;
    int p    = rem & 1;
    int nt   = frag >> 5;
    int kb   = frag & 31;
    int K    = kb * 32 + ((lane >= 16) ? 16 : 0) + 2 * j + p;
    int N    = nt * 16 + (lane & 15);
    wp[gid]  = f2bf(W[(size_t)N * WSTRIDE + K]);
}

// ---------------------------------------------------------------------------
// Kernel 2: base[b][h] = sum_d cur[b][d] * Wc[h][d] + b_proj[h]
// ---------------------------------------------------------------------------
__global__ void base_kernel(const float* __restrict__ cur, const float* __restrict__ W,
                            const float* __restrict__ bias, float* __restrict__ base) {
    int gid = blockIdx.x * 256 + threadIdx.x;           // 0 .. 32767
    int b = gid >> 10, h = gid & 1023;
    const float4* wr = (const float4*)(W + (size_t)h * WSTRIDE + HIDDEN);
    const float4* cr = (const float4*)(cur + (size_t)b * HIDDEN);
    float acc = bias[h];
    #pragma unroll 4
    for (int d4 = 0; d4 < HIDDEN / 4; ++d4) {
        float4 w = wr[d4];
        float4 c = cr[d4];
        acc += w.x * c.x + w.y * c.y + w.z * c.z + w.w * c.w;
    }
    base[gid] = acc;
}

// ---------------------------------------------------------------------------
// Kernel 3: fused GEMM + tanh + dot(w_score) -> raw scores.
// Grid: 1024 WGs, each owns 64 consecutive rows of (b,s). 8 wave32 / WG.
// LDS: 64-row A tile as bf16 (padded stride), base row, w_score, score buf.
// base/w_score staged via CDNA5 GLOBAL_LOAD_ASYNC_TO_LDS (ASYNCcnt DMA).
// ---------------------------------------------------------------------------
#define AST (HIDDEN + 8)   // padded LDS row stride in bf16 elements (2064 B)

__global__ __launch_bounds__(256, 1)
void score_kernel(const float* __restrict__ past, const unsigned short* __restrict__ wp,
                  const float* __restrict__ baseg, const float* __restrict__ wscore,
                  float* __restrict__ scores) {
    extern __shared__ char smem[];
    unsigned short* Abuf = (unsigned short*)smem;                         // 64*AST bf16
    float* baseS    = (float*)(smem + (size_t)64 * AST * sizeof(unsigned short));
    float* wsS      = baseS + HIDDEN;
    float* scoreBuf = wsS + HIDDEN;

    const int tid = threadIdx.x;
    const int m0  = blockIdx.x * 64;       // global flattened (b,s) row
    const int b   = m0 >> 11;              // / SEQ

    // Async DMA: stage base row + w_score into LDS (b128 per lane, ASYNCcnt).
    {
        unsigned ldsB = (unsigned)(uintptr_t)(baseS + tid * 4);
        unsigned ldsW = (unsigned)(uintptr_t)(wsS + tid * 4);
        unsigned long long gB = (unsigned long long)(uintptr_t)(baseg + (size_t)b * HIDDEN + tid * 4);
        unsigned long long gW = (unsigned long long)(uintptr_t)(wscore + tid * 4);
        asm volatile("global_load_async_to_lds_b128 %0, %1, off"
                     :: "v"(ldsB), "v"(gB) : "memory");
        asm volatile("global_load_async_to_lds_b128 %0, %1, off"
                     :: "v"(ldsW), "v"(gW) : "memory");
    }
    if (tid < 64) scoreBuf[tid] = 0.f;

    // Stage 64 x 1024 f32 rows of past -> bf16 LDS tile (coalesced float4).
    const float4* src = (const float4*)(past + (size_t)m0 * HIDDEN);
    for (int i = tid; i < 64 * (HIDDEN / 4); i += 256) {
        int row = i >> 8;                  // / (HIDDEN/4)
        int c4  = i & 255;
        float4 v = src[(size_t)row * (HIDDEN / 4) + c4];
        unsigned short* dst = Abuf + (size_t)row * AST + c4 * 4;
        dst[0] = f2bf(v.x); dst[1] = f2bf(v.y); dst[2] = f2bf(v.z); dst[3] = f2bf(v.w);
    }
    wait_asynccnt0();
    __syncthreads();

    const int wave = tid >> 5;             // 0..7 (wave32)
    const int lane = tid & 31;
    const int mi   = wave & 3;             // 16-row M subtile
    const int nh   = wave >> 2;            // N half (interleaved groups)
    const int mloc = mi * 16 + (lane & 15);
    const int kHalf = (lane >= 16) ? 16 : 0;

    const unsigned short* arow = Abuf + (size_t)mloc * AST;

    float p[8];
    #pragma unroll
    for (int r = 0; r < 8; ++r) p[r] = 0.f;

    // 64 N-tiles total; each wave takes 8 groups of 4 tiles.
    for (int g = nh; g < 16; g += 2) {
        const int nt0 = g * 4;
        v8f cacc[4];
        #pragma unroll
        for (int t = 0; t < 4; ++t) {
            v8f z = {0.f, 0.f, 0.f, 0.f, 0.f, 0.f, 0.f, 0.f};
            cacc[t] = z;
        }

        for (int kb = 0; kb < 32; ++kb) {
            // A fragment: 2x b128 from LDS (contiguous 16-K half per lane half, same order as packed B).
            union { v16bf v; uint4 q[2]; } au;
            const uint4* ap = (const uint4*)(arow + kb * 32 + kHalf);
            au.q[0] = ap[0];
            au.q[1] = ap[1];               // +16B = K offset +8

            #pragma unroll
            for (int t = 0; t < 4; ++t) {
                const int f = (nt0 + t) * 32 + kb;
                const uint4* bp = (const uint4*)wp + (size_t)f * 64 + lane * 2;
                union { v16bf v; uint4 q[2]; } bu;
                bu.q[0] = bp[0];
                bu.q[1] = bp[1];
                cacc[t] = __builtin_amdgcn_wmma_f32_16x16x32_bf16(
                    false, au.v, false, bu.v, (short)0, cacc[t], false, false);
            }
        }

        // Epilogue: per lane (column n), p[r] += w[n] * tanh(c[r] + base[n]).
        const int ncol = lane & 15;
        #pragma unroll
        for (int t = 0; t < 4; ++t) {
            const int n = (nt0 + t) * 16 + ncol;
            const float bb = baseS[n];
            const float ww = wsS[n];
            #pragma unroll
            for (int r = 0; r < 8; ++r)
                p[r] += ww * fast_tanhf(cacc[t][r] + bb);
        }
    }

    // Reduce over the 16 lanes holding the same row half, combine waves in LDS.
    #pragma unroll
    for (int r = 0; r < 8; ++r) {
        float v = p[r];
        v += __shfl_xor(v, 1, 32);
        v += __shfl_xor(v, 2, 32);
        v += __shfl_xor(v, 4, 32);
        v += __shfl_xor(v, 8, 32);
        p[r] = v;
    }
    if ((lane & 15) == 0) {
        const int rowBase = mi * 16 + ((lane >= 16) ? 8 : 0);
        #pragma unroll
        for (int r = 0; r < 8; ++r)
            atomicAdd(&scoreBuf[rowBase + r], p[r]);
    }
    __syncthreads();
    if (tid < 64) scores[(size_t)m0 + tid] = scoreBuf[tid];
}

// ---------------------------------------------------------------------------
// Kernel 4: in-place softmax over S per batch row.
// ---------------------------------------------------------------------------
__global__ void softmax_kernel(float* __restrict__ attn) {
    __shared__ float rbuf[8];
    __shared__ float sbuf[8];
    const int b = blockIdx.x, tid = threadIdx.x;
    float* row = attn + (size_t)b * SEQ;

    float vals[8];
    float lmax = -INFINITY;
    #pragma unroll
    for (int i = 0; i < 8; ++i) {
        vals[i] = row[tid + i * 256];
        lmax = fmaxf(lmax, vals[i]);
    }
    #pragma unroll
    for (int off = 16; off; off >>= 1) lmax = fmaxf(lmax, __shfl_xor(lmax, off, 32));
    if ((tid & 31) == 0) rbuf[tid >> 5] = lmax;
    __syncthreads();
    float gmax = rbuf[0];
    #pragma unroll
    for (int i = 1; i < 8; ++i) gmax = fmaxf(gmax, rbuf[i]);

    float lsum = 0.f;
    #pragma unroll
    for (int i = 0; i < 8; ++i) { vals[i] = __expf(vals[i] - gmax); lsum += vals[i]; }
    #pragma unroll
    for (int off = 16; off; off >>= 1) lsum += __shfl_xor(lsum, off, 32);
    if ((tid & 31) == 0) sbuf[tid >> 5] = lsum;
    __syncthreads();
    float gsum = 0.f;
    #pragma unroll
    for (int i = 0; i < 8; ++i) gsum += sbuf[i];

    const float inv = 1.f / gsum;
    #pragma unroll
    for (int i = 0; i < 8; ++i) row[tid + i * 256] = vals[i] * inv;
}

// ---------------------------------------------------------------------------
// Kernel 5: attended[b][d] = sum_s attn[b][s] * past[b][s][d]
// ---------------------------------------------------------------------------
__global__ void attend_kernel(const float* __restrict__ past, const float* __restrict__ attn,
                              float* __restrict__ out) {
    const int b = blockIdx.y;
    const int d = blockIdx.x * 256 + threadIdx.x;
    const float* prow = past + (size_t)b * SEQ * HIDDEN + d;
    const float* arow = attn + (size_t)b * SEQ;
    float acc = 0.f;
    #pragma unroll 8
    for (int s = 0; s < SEQ; ++s)
        acc += arow[s] * prow[(size_t)s * HIDDEN];
    out[(size_t)b * HIDDEN + d] = acc;
}

// ---------------------------------------------------------------------------
extern "C" void kernel_launch(void* const* d_in, const int* in_sizes, int n_in,
                              void* d_out, int out_size, void* d_ws, size_t ws_size,
                              hipStream_t stream) {
    const float* cur    = (const float*)d_in[0];   // (B, H)
    const float* past   = (const float*)d_in[1];   // (B, S, H)
    const float* Wproj  = (const float*)d_in[2];   // (H, 2H)
    const float* bproj  = (const float*)d_in[3];   // (H,)
    const float* wscore = (const float*)d_in[4];   // (H,)

    float* attended = (float*)d_out;                       // (B, H)
    float* attn     = attended + BATCH * HIDDEN;           // (B, S)

    unsigned short* wp = (unsigned short*)d_ws;            // 2 MB packed bf16 Wp
    float* base = (float*)((char*)d_ws + (size_t)HIDDEN * HIDDEN * sizeof(unsigned short));

    pack_wp_kernel<<<(HIDDEN * HIDDEN) / 256, 256, 0, stream>>>(Wproj, wp);
    base_kernel<<<(BATCH * HIDDEN) / 256, 256, 0, stream>>>(cur, Wproj, bproj, base);

    size_t smem = (size_t)64 * AST * sizeof(unsigned short) +
                  (2 * HIDDEN + 64) * sizeof(float);
    score_kernel<<<(BATCH * SEQ) / 64, 256, smem, stream>>>(past, wp, base, wscore, attn);

    softmax_kernel<<<BATCH, 256, 0, stream>>>(attn);
    attend_kernel<<<dim3(HIDDEN / 256, BATCH), 256, 0, stream>>>(past, attn, attended);
}
